// MultiViewAttention_90795608637859
// MI455X (gfx1250) — matrix operand order repaired and checked by reading, not compiled, not run on hardware
//
#include <hip/hip_runtime.h>
#include <stdint.h>

// ---------------------------------------------------------------------------
// MI455X (gfx1250) channel-attention block:
//   qkv  = W_qkv @ x + b_qkv                     (bf16 WMMA GEMM, f32 accum)
//   attn = softmax(q k^T / sqrt(32)) per (b,h)   (f32 WMMA 16x16x4, K=9216)
//   Weff = blockdiag-fold of attn into W_proj    (tiny VALU kernel)
//   out  = Weff @ v + b_proj                     (bf16 WMMA GEMM, f32 accum)
// Memory-bound (~1.2 GB HBM @ 23.3 TB/s). Double-buffered LDS everywhere;
// attention staging uses GLOBAL_LOAD_ASYNC_TO_LDS_B128 when available.
// ---------------------------------------------------------------------------

typedef __attribute__((ext_vector_type(16))) __bf16 v16bf;
typedef __attribute__((ext_vector_type(8)))  float  v8f;
typedef __attribute__((ext_vector_type(2)))  float  v2f;
typedef __attribute__((ext_vector_type(4)))  int    v4i;

union Frag16 { v16bf v; uint4 q[2]; };

#if defined(__HIP_DEVICE_COMPILE__) && defined(__gfx1250__) && \
    __has_builtin(__builtin_amdgcn_global_load_async_to_lds_b128)
#define HAVE_ASYNC_LDS 1
#else
#define HAVE_ASYNC_LDS 0
#endif

#if HAVE_ASYNC_LDS
#if __has_builtin(__builtin_amdgcn_s_wait_asynccnt)
#define WAIT_ASYNC(n) __builtin_amdgcn_s_wait_asynccnt(n)
#else
#define WAIT_ASYNC(n) asm volatile("s_wait_asynccnt %0" ::"i"(n) : "memory")
#endif
// probe-confirmed signature: (int4* gsrc, int4 as3* ldst, imm offset, imm cpol)
#define ASYNC_CP16(gsrc, ldst)                                            \
  __builtin_amdgcn_global_load_async_to_lds_b128(                         \
      (v4i*)(gsrc),                                                       \
      (__attribute__((address_space(3))) v4i*)(ldst), 0, 0)
#endif

__device__ __forceinline__ unsigned short f32_to_bf16_rne(float f) {
  unsigned int u = __float_as_uint(f);
  u += 0x7FFFu + ((u >> 16) & 1u);          // round-to-nearest-even
  return (unsigned short)(u >> 16);
}
__device__ __forceinline__ unsigned int pack_bf16x2(float lo, float hi) {
  return (unsigned int)f32_to_bf16_rne(lo) |
         ((unsigned int)f32_to_bf16_rne(hi) << 16);
}

// ---------------------------------------------------------------------------
// GEMM: C[b] = A[b](MxK, rm) * B[b](KxN, rm) + bias[M]
// BM=BN=128, BK=32; 8 waves, wave grid 4(M)x2(N), 2x4 fragments/wave of
// v_wmma_f32_16x16x32_bf16. Double-buffered LDS, register-staged loads,
// one barrier per K step. All dims divide tiles exactly (768/256/9216).
// ---------------------------------------------------------------------------
#define BM 128
#define BN 128
#define BK 32

struct StageRegs { float4 a[4]; float4 b[4]; };

__device__ __forceinline__ void stage_load(const float* __restrict__ A,
                                           const float* __restrict__ B,
                                           int N, int K, int bm, int bn, int k0,
                                           int tid, StageRegs& r) {
  #pragma unroll
  for (int i = 0; i < 4; ++i) {               // A: 128x32, 8 float4 per row
    const int c = tid + i * 256;
    const int row = c >> 3, col = (c & 7) * 4;
    r.a[i] = *(const float4*)(A + (size_t)(bm + row) * K + k0 + col);
  }
  #pragma unroll
  for (int i = 0; i < 4; ++i) {               // B: 32x128, 32 float4 per k-row
    const int c = tid + i * 256;
    const int krow = c >> 5, ncol = (c & 31) * 4;
    r.b[i] = *(const float4*)(B + (size_t)(k0 + krow) * N + bn + ncol);
  }
}

__device__ __forceinline__ void stage_store(unsigned short* sA, unsigned short* sB,
                                            int tid, const StageRegs& r) {
  #pragma unroll
  for (int i = 0; i < 4; ++i) {               // A tile bf16, row-major [m][k]
    const int c = tid + i * 256;
    const int row = c >> 3, col = (c & 7) * 4;
    uint2 p;
    p.x = pack_bf16x2(r.a[i].x, r.a[i].y);
    p.y = pack_bf16x2(r.a[i].z, r.a[i].w);
    *(uint2*)&sA[row * BK + col] = p;
  }
  #pragma unroll
  for (int i = 0; i < 4; ++i) {               // B tile bf16, transposed [n][k]
    const int c = tid + i * 256;
    const int krow = c >> 5, ncol = (c & 31) * 4;
    sB[(ncol + 0) * BK + krow] = f32_to_bf16_rne(r.b[i].x);
    sB[(ncol + 1) * BK + krow] = f32_to_bf16_rne(r.b[i].y);
    sB[(ncol + 2) * BK + krow] = f32_to_bf16_rne(r.b[i].z);
    sB[(ncol + 3) * BK + krow] = f32_to_bf16_rne(r.b[i].w);
  }
}

__global__ __launch_bounds__(256)
void gemm_wmma_bf16(const float* __restrict__ Ab, long long strideA,
                    const float* __restrict__ Bb, long long strideB,
                    float* __restrict__ Cb, long long strideC,
                    const float* __restrict__ bias,
                    int M, int N, int K)
{
  const float* A = Ab + (size_t)blockIdx.z * strideA;
  const float* B = Bb + (size_t)blockIdx.z * strideB;
  float*       C = Cb + (size_t)blockIdx.z * strideC;

  const int bm    = blockIdx.y * BM;
  const int bn    = blockIdx.x * BN;
  const int tid   = threadIdx.x;
  const int wave  = tid >> 5;
  const int lane  = tid & 31;
  const int l15   = lane & 15;
  const int lhalf = lane >> 4;

  const int wm = (wave & 3) * 32;
  const int wn = (wave >> 2) * 64;

  __shared__ unsigned short smA[2][BM * BK];
  __shared__ unsigned short smB[2][BN * BK];

  v8f acc[2][4];
  #pragma unroll
  for (int mi = 0; mi < 2; ++mi)
    #pragma unroll
    for (int ni = 0; ni < 4; ++ni)
      acc[mi][ni] = (v8f)0.0f;

  StageRegs r;
  stage_load(A, B, N, K, bm, bn, 0, tid, r);
  stage_store(smA[0], smB[0], tid, r);
  __syncthreads();

  int buf = 0;
  for (int k0 = 0; k0 < K; k0 += BK) {
    const bool has_next = (k0 + BK) < K;
    if (has_next) stage_load(A, B, N, K, bm, bn, k0 + BK, tid, r);

    const unsigned short* sA = smA[buf];
    const unsigned short* sB = smB[buf];
    Frag16 af[2], bf[4];
    #pragma unroll
    for (int mi = 0; mi < 2; ++mi) {
      const int base = (wm + mi * 16 + l15) * BK + lhalf * 8;
      af[mi].q[0] = *(const uint4*)&sA[base];
      af[mi].q[1] = *(const uint4*)&sA[base + 16];
    }
    #pragma unroll
    for (int ni = 0; ni < 4; ++ni) {
      const int base = (wn + ni * 16 + l15) * BK + lhalf * 8;
      bf[ni].q[0] = *(const uint4*)&sB[base];
      bf[ni].q[1] = *(const uint4*)&sB[base + 16];
    }
    #pragma unroll
    for (int mi = 0; mi < 2; ++mi)
      #pragma unroll
      for (int ni = 0; ni < 4; ++ni)
        acc[mi][ni] = __builtin_amdgcn_wmma_f32_16x16x32_bf16(
            false, af[mi].v, false, bf[ni].v, (short)0, acc[mi][ni], false, false);

    if (has_next) stage_store(smA[buf ^ 1], smB[buf ^ 1], tid, r);
    __syncthreads();
    buf ^= 1;
  }

  #pragma unroll
  for (int mi = 0; mi < 2; ++mi)
    #pragma unroll
    for (int ni = 0; ni < 4; ++ni) {
      const int col = bn + wn + ni * 16 + l15;
      #pragma unroll
      for (int r8 = 0; r8 < 8; ++r8) {
        const int row = bm + wm + mi * 16 + lhalf * 8 + r8;
        C[(size_t)row * N + col] = acc[mi][ni][r8] + bias[row];
      }
    }
}

// ---------------------------------------------------------------------------
// attn[b,h] = softmax( (q k^T) / sqrt(32) ), q,k = 32x9216 rows of qkv.
// 128-wide K chunks double-buffered in LDS; chunk c+1 streams in via
// GLOBAL_LOAD_ASYNC_TO_LDS_B128 (ASYNCcnt) while chunk c feeds
// v_wmma_f32_16x16x4_f32 (exact f32). 8-wave partials reduced in LDS.
// ---------------------------------------------------------------------------
__device__ __forceinline__ void attn_issue_chunk(const float* __restrict__ Q,
                                                 const float* __restrict__ Kp,
                                                 int c0, int tid,
                                                 float* qs, float* ks) {
  #pragma unroll
  for (int i = 0; i < 4; ++i) {
    const int c = tid + i * 256;
    const int row = c >> 5, col = (c & 31) * 4;
    const size_t go = (size_t)row * 9216 + c0 + col;
    const int    lo = row * 128 + col;
#if HAVE_ASYNC_LDS
    ASYNC_CP16(Q + go, &qs[lo]);
    ASYNC_CP16(Kp + go, &ks[lo]);
#else
    *(float4*)&qs[lo] = *(const float4*)(Q + go);
    *(float4*)&ks[lo] = *(const float4*)(Kp + go);
#endif
  }
}

__global__ __launch_bounds__(256)
void attn_softmax_kernel(const float* __restrict__ qkv, float* __restrict__ attn)
{
  const int h = blockIdx.x, b = blockIdx.y;
  const int tid   = threadIdx.x;
  const int wave  = tid >> 5;
  const int lane  = tid & 31;
  const int l15   = lane & 15;
  const int lhalf = lane >> 4;

  const float* Q  = qkv + ((size_t)b * 768 +       h * 32) * 9216;
  const float* Kp = qkv + ((size_t)b * 768 + 256 + h * 32) * 9216;

  __shared__ float Qs[2][32 * 128];
  __shared__ float Ks[2][32 * 128];
  __shared__ float Pb[8 * 32 * 32];

  v8f acc[2][2];
  #pragma unroll
  for (int mi = 0; mi < 2; ++mi)
    #pragma unroll
    for (int ei = 0; ei < 2; ++ei) acc[mi][ei] = (v8f)0.0f;

  attn_issue_chunk(Q, Kp, 0, tid, Qs[0], Ks[0]);

  int buf = 0;
  for (int c0 = 0; c0 < 9216; c0 += 128) {
    if (c0 + 128 < 9216) {
      attn_issue_chunk(Q, Kp, c0 + 128, tid, Qs[buf ^ 1], Ks[buf ^ 1]);
#if HAVE_ASYNC_LDS
      WAIT_ASYNC(8);   // 8 just-issued copies may remain; current chunk done
#endif
    } else {
#if HAVE_ASYNC_LDS
      WAIT_ASYNC(0);
#endif
    }
    __syncthreads();   // current chunk visible to all waves

    const float* qs = Qs[buf];
    const float* ks = Ks[buf];
    #pragma unroll
    for (int s = 0; s < 4; ++s) {
      const int kk = (wave * 4 + s) * 4 + lhalf * 2;
      v2f af[2], bfr[2];
      #pragma unroll
      for (int mi = 0; mi < 2; ++mi)
        af[mi] = *(const v2f*)&qs[(mi * 16 + l15) * 128 + kk];
      #pragma unroll
      for (int ei = 0; ei < 2; ++ei)
        bfr[ei] = *(const v2f*)&ks[(ei * 16 + l15) * 128 + kk];
      #pragma unroll
      for (int mi = 0; mi < 2; ++mi)
        #pragma unroll
        for (int ei = 0; ei < 2; ++ei)
          acc[mi][ei] = __builtin_amdgcn_wmma_f32_16x16x4_f32(
              false, af[mi], false, bfr[ei], (short)0, acc[mi][ei], false, false);
    }
    __syncthreads();   // all reads of buf done before it is refilled
    buf ^= 1;
  }

  // per-wave 32x32 partials -> LDS
  float* pw = &Pb[wave * 1024];
  #pragma unroll
  for (int mi = 0; mi < 2; ++mi)
    #pragma unroll
    for (int ei = 0; ei < 2; ++ei)
      #pragma unroll
      for (int r8 = 0; r8 < 8; ++r8)
        pw[(mi * 16 + lhalf * 8 + r8) * 32 + ei * 16 + l15] = acc[mi][ei][r8];
  __syncthreads();

  // cross-wave reduce + scale (reuse Qs[0] as the 32x32 logits buffer)
  float* Ls = Qs[0];
  const float scale = 0.17677669529663687f;   // 32^-0.5
  #pragma unroll
  for (int i = 0; i < 4; ++i) {
    const int idx = tid + i * 256;
    float s = 0.0f;
    #pragma unroll
    for (int w = 0; w < 8; ++w) s += Pb[w * 1024 + idx];
    Ls[idx] = s * scale;
  }
  __syncthreads();

  // row softmax (32 rows, one lane each) and store
  if (tid < 32) {
    float* rowp = &Ls[tid * 32];
    float m = rowp[0];
    #pragma unroll
    for (int e = 1; e < 32; ++e) m = fmaxf(m, rowp[e]);
    float sum = 0.0f;
    #pragma unroll
    for (int e = 0; e < 32; ++e) { float ex = __expf(rowp[e] - m); rowp[e] = ex; sum += ex; }
    const float inv = 1.0f / sum;
    float* dst = attn + (((size_t)b * 8 + h) * 32 + tid) * 32;
    #pragma unroll
    for (int e = 0; e < 32; ++e) dst[e] = rowp[e] * inv;
  }
}

// ---------------------------------------------------------------------------
// Weff[b][o][h*32+e] = sum_d W_proj[o][h*32+d] * attn[b,h,d,e]
// ---------------------------------------------------------------------------
__global__ __launch_bounds__(256)
void weff_kernel(const float* __restrict__ attn, const float* __restrict__ w_proj,
                 float* __restrict__ weff)
{
  const int h = blockIdx.x, b = blockIdx.y;
  const int tid = threadIdx.x;
  __shared__ float As[1024];                    // attn[b,h] 32x32
  *(float4*)&As[tid * 4] = *(const float4*)(attn + ((size_t)b * 8 + h) * 1024 + tid * 4);
  __syncthreads();

  const float* wp = w_proj + (size_t)tid * 256 + h * 32;
  float acc[32];
  #pragma unroll
  for (int e = 0; e < 32; ++e) acc[e] = 0.0f;
  #pragma unroll 4
  for (int d = 0; d < 32; ++d) {
    const float wv = wp[d];
    const float* ar = &As[d * 32];
    #pragma unroll
    for (int e = 0; e < 32; ++e) acc[e] = fmaf(wv, ar[e], acc[e]);
  }
  float* dst = weff + (size_t)b * 65536 + (size_t)tid * 256 + h * 32;
  #pragma unroll
  for (int e = 0; e < 32; ++e) dst[e] = acc[e];
}

// ---------------------------------------------------------------------------
extern "C" void kernel_launch(void* const* d_in, const int* in_sizes, int n_in,
                              void* d_out, int out_size, void* d_ws, size_t ws_size,
                              hipStream_t stream) {
  const float* x      = (const float*)d_in[0];   // (16,256,96,96)
  const float* w_qkv  = (const float*)d_in[1];   // (768,256)
  const float* b_qkv  = (const float*)d_in[2];   // (768)
  const float* w_proj = (const float*)d_in[3];   // (256,256)
  const float* b_proj = (const float*)d_in[4];   // (256)
  float* out = (float*)d_out;                    // (16,256,96,96)

  float* ws   = (float*)d_ws;
  float* qkv  = ws;                                   // 16*768*9216 f32
  float* attn = qkv  + (size_t)16 * 768 * 9216;       // 16*8*32*32   f32
  float* weff = attn + (size_t)16 * 8 * 1024;         // 16*256*256   f32

  // 1) QKV projection: per batch  qkv[b] = w_qkv @ x[b] + b_qkv
  dim3 g1(9216 / BN, 768 / BM, 16);
  gemm_wmma_bf16<<<g1, 256, 0, stream>>>(
      w_qkv, 0LL,
      x, (long long)256 * 9216,
      qkv, (long long)768 * 9216,
      b_qkv, 768, 9216, 256);

  // 2) 32x32 channel attention + softmax per (b,h)
  attn_softmax_kernel<<<dim3(8, 16), 256, 0, stream>>>(qkv, attn);

  // 3) fold attention into projection weights
  weff_kernel<<<dim3(8, 16), 256, 0, stream>>>(attn, w_proj, weff);

  // 4) out[b] = Weff[b] @ v[b] + b_proj   (v = qkv rows 512..767)
  dim3 g2(9216 / BN, 256 / BM, 16);
  gemm_wmma_bf16<<<g2, 256, 0, stream>>>(
      weff, (long long)256 * 256,
      qkv + (size_t)512 * 9216, (long long)768 * 9216,
      out, (long long)256 * 9216,
      b_proj, 256, 9216, 256);
}
